// FactorizationTTMLinear_22351009808660
// MI455X (gfx1250) — matrix operand order repaired and checked by reading, not compiled, not run on hardware
//
#include <hip/hip_runtime.h>

// ---------------------------------------------------------------------------
// TTM linear (3 fused GEMMs per token) for gfx1250 using v_wmma_f32_16x16x32_bf16,
// with double-buffered global_load_async_to_lds prefetch of the next token's x.
// Per token: G1 (256x16 * 16x128), G2 (256x128 * 128x128), G3 (256x128 * 128x16),
// all intermediates kept in LDS as bf16 in the layout required by the next GEMM.
// ---------------------------------------------------------------------------

typedef __attribute__((ext_vector_type(16))) __bf16 v16bf;
typedef __attribute__((ext_vector_type(8)))  float  v8f;
typedef __attribute__((ext_vector_type(4)))  __bf16 bf16x4;

union FragU {
    uint4 q[2];
    v16bf v;
};

// Strides in elements (padded to avoid LDS bank conflicts; byte strides are
// multiples of 16 so uint4 LDS loads stay aligned).
#define STRIDE_K32  40    // tiles with K<=32 (80 B  = 20 banks)
#define STRIDE_K128 136   // tiles with K=128 (272 B = 68 banks)

// LDS layout (bytes). NOTE: no static __shared__ in this TU, so the dynamic
// LDS block starts at LDS byte offset 0 and these offsets are raw LDS
// addresses usable as VDST for global_load_async_to_lds.
#define OFF_B2  0                                   // 128 x 136 bf16 = 34816
#define OFF_B1  (OFF_B2 + 128 * STRIDE_K128 * 2)    // 128 x  40 bf16 = 10240
#define OFF_B3  (OFF_B1 + 128 * STRIDE_K32  * 2)    //  16 x 136 bf16 =  4352
#define OFF_X   (OFF_B3 + 16  * STRIDE_K128 * 2)    // 256 x  40 bf16 = 20480
#define OFF_Z1  (OFF_X  + 256 * STRIDE_K32  * 2)    // 256 x 136 bf16 = 69632
#define OFF_Z2  (OFF_Z1 + 256 * STRIDE_K128 * 2)    // 256 x 136 bf16 = 69632
#define OFF_XS0 (OFF_Z2 + 256 * STRIDE_K128 * 2)    // 4096 f32 stage = 16384
#define OFF_XS1 (OFF_XS0 + 4096 * 4)                // 4096 f32 stage = 16384
#define SMEM_BYTES (OFF_XS1 + 4096 * 4)             // 241920 total

// A fragment (16x32 bf16, MxK): lane holds row m = lane&15;
// VGPR elems 0..7 -> K = 8h+0..7, elems 8..15 -> K = 16+8h+0..7 (h = lane>>4).
__device__ __forceinline__ v16bf load_a(const __bf16* tile, int stride, int lane) {
    const int m = lane & 15, h = lane >> 4;
    const __bf16* r = tile + m * stride + 8 * h;
    FragU f;
    f.q[0] = *(const uint4*)(r);
    f.q[1] = *(const uint4*)(r + 16);
    return f.v;
}

// B fragment (32x16 bf16, KxN) read from N-major (transposed) storage:
// lane holds column n = lane&15; elems 0..7 -> K = 8h+0..7, elems 8..15 -> K = 16+8h+0..7.
__device__ __forceinline__ v16bf load_b(const __bf16* tileT, int stride, int lane) {
    const int n = lane & 15, h = lane >> 4;
    const __bf16* r = tileT + n * stride + 8 * h;
    FragU f;
    f.q[0] = *(const uint4*)(r);
    f.q[1] = *(const uint4*)(r + 16);
    return f.v;
}

__device__ __forceinline__ v8f wmma_bf16(v16bf a, v16bf b, v8f c) {
    return __builtin_amdgcn_wmma_f32_16x16x32_bf16(false, a, false, b,
                                                   (short)0, c, false, false);
}

// Async-copy one token's x row (4096 f32 = 16 KB) into an LDS staging buffer.
// 256 threads x 4 x b128. Tracked by ASYNCcnt.
__device__ __forceinline__ void async_stage_x(const float* gsrc, unsigned ldsByteOff,
                                              int tid) {
    unsigned long long g = (unsigned long long)(uintptr_t)gsrc
                         + (unsigned long long)tid * 16u;
    unsigned l = ldsByteOff + (unsigned)tid * 16u;
    #pragma unroll
    for (int i = 0; i < 4; ++i) {
        asm volatile("global_load_async_to_lds_b128 %0, %1, off"
                     :: "v"(l), "v"(g) : "memory");
        g += 256u * 16u;
        l += 256u * 16u;
    }
}

__device__ __forceinline__ void wait_async0() {
    asm volatile("s_wait_asynccnt 0x0" ::: "memory");
}

__global__ __launch_bounds__(256)
void ttm_linear_kernel(const float* __restrict__ x, const float* __restrict__ bias,
                       const float* __restrict__ core0, const float* __restrict__ core1,
                       const float* __restrict__ core2, float* __restrict__ out,
                       int tokensPerWG, int nTokens) {
    extern __shared__ char smem[];
    __bf16* B2 = (__bf16*)(smem + OFF_B2);   // [(n2*8+p2)][(p1*16+m2)]
    __bf16* B1 = (__bf16*)(smem + OFF_B1);   // [(n1*8+p1)][m1]  (K padded w/ zeros)
    __bf16* B3 = (__bf16*)(smem + OFF_B3);   // [n3][(p2*16+m3)]
    __bf16* X  = (__bf16*)(smem + OFF_X);    // [(m2*16+m3)][m1] (K padded w/ zeros)
    __bf16* Z1 = (__bf16*)(smem + OFF_Z1);   // [(m3*16+n1)][(p1*16+m2)]
    __bf16* Z2 = (__bf16*)(smem + OFF_Z2);   // [(n1*16+n2)][(p2*16+m3)]

    const int tid  = threadIdx.x;
    const int lane = tid & 31;
    // Wave index made provably wave-uniform -> scalar loop control, and WMMA
    // executes with EXEC all-ones structurally.
    const int u = __builtin_amdgcn_readfirstlane(tid >> 5);   // 0..7, col tile

    // ---- prologue: start async staging of this WG's first token ----
    {
        const int tok0 = blockIdx.x * tokensPerWG;
        if (tok0 < nTokens)
            async_stage_x(x + (size_t)tok0 * 4096, OFF_XS0, tid);
    }

    // ---- one-time per-WG core preparation (fp32 -> bf16, transposed layouts) ----
    // core1 (8,16,16,8): flat = p1*2048 + m2*128 + n2*8 + p2
    for (int i = tid; i < 128 * 128; i += 256) {
        int row = i >> 7, k = i & 127;  // row=(n2*8+p2), k=(p1*16+m2)
        B2[row * STRIDE_K128 + k] = (__bf16)core1[(k >> 4) * 2048 + (k & 15) * 128 + row];
    }
    // core0 (1,16,16,8): flat = m1*128 + n1*8 + p1 ; zero-pad K 16..39
    for (int i = tid; i < 128 * STRIDE_K32; i += 256) {
        int row = i / STRIDE_K32, k = i % STRIDE_K32;  // row=(n1*8+p1), k=m1
        float v = (k < 16) ? core0[k * 128 + row] : 0.0f;
        B1[row * STRIDE_K32 + k] = (__bf16)v;
    }
    // core2 (8,16,16,1): flat = p2*256 + m3*16 + n3
    for (int i = tid; i < 16 * 128; i += 256) {
        int row = i >> 7, k = i & 127;  // row=n3, k=(p2*16+m3)
        B3[row * STRIDE_K128 + k] = (__bf16)core2[(k >> 4) * 256 + (k & 15) * 16 + row];
    }
    // zero the K-pad region of X once (K in [16,40))
    for (int i = tid; i < 256 * 24; i += 256) {
        int row = i / 24, k = 16 + i % 24;
        X[row * STRIDE_K32 + k] = (__bf16)0.0f;
    }

    for (int it = 0; it < tokensPerWG; ++it) {
        const int tok = blockIdx.x * tokensPerWG + it;
        if (tok >= nTokens) break;

        const float* curS = (const float*)(smem + ((it & 1) ? OFF_XS1 : OFF_XS0));
        wait_async0();            // my async writes into curS have landed
        __syncthreads();          // everyone's landed; prev iteration fully done

        // kick off prefetch of next token into the other stage buffer
        const int ntok = tok + 1;
        if ((it + 1) < tokensPerWG && ntok < nTokens)
            async_stage_x(x + (size_t)ntok * 4096,
                          (it & 1) ? OFF_XS0 : OFF_XS1, tid);

        // ---- stage X: curS[m1*256 + m2*16+m3] -> X[(m2*16+m3)][m1] (bf16) ----
        #pragma unroll
        for (int i = 0; i < 16; ++i)
            X[tid * STRIDE_K32 + i] = (__bf16)curS[i * 256 + tid];
        __syncthreads();

        const int N = lane & 15, h = lane >> 4;
        const int kA0 = (N & 7) * 16;               // K-index base (+ t later)

        // ---- G1: Z1[(m2,m3)][(n1,p1)] = sum_m1 X * core0 ; store permuted -> A2 ----
        {
            const v16bf b = load_b(B1 + (u * 16) * STRIDE_K32, STRIDE_K32, lane);
            const int n1 = 2 * u + (N >> 3);
            for (int t = 0; t < 16; t += 4) {       // four row tiles share b
                v16bf a[4];
                #pragma unroll
                for (int q = 0; q < 4; ++q)
                    a[q] = load_a(X + (t + q) * 16 * STRIDE_K32, STRIDE_K32, lane);
                v8f c[4] = {};
                #pragma unroll
                for (int q = 0; q < 4; ++q)
                    c[q] = wmma_bf16(a[q], b, c[q]);
                #pragma unroll
                for (int j = 0; j < 8; ++j) {
                    const int m3 = j + 8 * h;
                    // 4 consecutive K entries (m2 = t..t+3): one packed b64 store
                    bf16x4 pk = { (__bf16)c[0][j], (__bf16)c[1][j],
                                  (__bf16)c[2][j], (__bf16)c[3][j] };
                    *(bf16x4*)(Z1 + (m3 * 16 + n1) * STRIDE_K128 + kA0 + t) = pk;
                }
            }
        }
        __syncthreads();

        // ---- G2: Z2[(m3,n1)][(n2,p2)] = sum_(p1,m2) Z1 * core1 ; store -> A3 ----
        {
            v16bf bfr[4];
            #pragma unroll
            for (int kc = 0; kc < 4; ++kc)
                bfr[kc] = load_b(B2 + (u * 16) * STRIDE_K128 + kc * 32,
                                 STRIDE_K128, lane);
            const int n2 = 2 * u + (N >> 3);
            for (int t = 0; t < 16; t += 4) {       // four row tiles share bfr[]
                v16bf aP[4], aQ[4];
                #pragma unroll
                for (int q = 0; q < 4; ++q)
                    aP[q] = load_a(Z1 + (t + q) * 16 * STRIDE_K128,
                                   STRIDE_K128, lane);
                v8f c[4] = {};
                #pragma unroll
                for (int kc = 0; kc < 4; ++kc) {
                    v16bf* cur = (kc & 1) ? aQ : aP;
                    v16bf* nxt = (kc & 1) ? aP : aQ;
                    if (kc < 3) {                   // prefetch next K-chunk's A
                        #pragma unroll
                        for (int q = 0; q < 4; ++q)
                            nxt[q] = load_a(Z1 + (t + q) * 16 * STRIDE_K128
                                               + (kc + 1) * 32,
                                            STRIDE_K128, lane);
                    }
                    #pragma unroll
                    for (int q = 0; q < 4; ++q)
                        c[q] = wmma_bf16(cur[q], bfr[kc], c[q]);
                }
                #pragma unroll
                for (int j = 0; j < 8; ++j) {
                    const int n1 = j + 8 * h;
                    // 4 consecutive K entries (m3 = t..t+3): one packed b64 store
                    bf16x4 pk = { (__bf16)c[0][j], (__bf16)c[1][j],
                                  (__bf16)c[2][j], (__bf16)c[3][j] };
                    *(bf16x4*)(Z2 + (n1 * 16 + n2) * STRIDE_K128 + kA0 + t) = pk;
                }
            }
        }
        __syncthreads();

        // ---- G3: y[(n1,n2)][n3] = sum_(p2,m3) Z2 * core2 ; + bias, fp32 out ----
        {
            float* yt = out + (size_t)tok * 4096;
            const int t0 = u, t1 = u + 8;           // two row tiles per wave
            v16bf b3[4], a0[4], a1[4];
            #pragma unroll
            for (int kc = 0; kc < 4; ++kc) {
                b3[kc] = load_b(B3 + kc * 32, STRIDE_K128, lane);
                a0[kc] = load_a(Z2 + t0 * 16 * STRIDE_K128 + kc * 32,
                                STRIDE_K128, lane);
                a1[kc] = load_a(Z2 + t1 * 16 * STRIDE_K128 + kc * 32,
                                STRIDE_K128, lane);
            }
            v8f c0 = {}, c1 = {};
            #pragma unroll
            for (int kc = 0; kc < 4; ++kc) {
                c0 = wmma_bf16(a0[kc], b3[kc], c0);
                c1 = wmma_bf16(a1[kc], b3[kc], c1);
            }
            #pragma unroll
            for (int j = 0; j < 8; ++j) {
                const int r0 = (t0 * 16 + j + 8 * h) * 16 + N;  // n1*256+n2*16+n3
                const int r1 = (t1 * 16 + j + 8 * h) * 16 + N;
                yt[r0] = c0[j] + bias[r0];
                yt[r1] = c1[j] + bias[r1];
            }
        }
        // no trailing barrier needed: next iteration starts with
        // s_wait_asynccnt + s_barrier before any LDS region is overwritten.
    }
}

extern "C" void kernel_launch(void* const* d_in, const int* in_sizes, int n_in,
                              void* d_out, int out_size, void* d_ws, size_t ws_size,
                              hipStream_t stream) {
    const float* x     = (const float*)d_in[0];
    const float* bias  = (const float*)d_in[1];
    const float* core0 = (const float*)d_in[2];
    const float* core1 = (const float*)d_in[3];
    const float* core2 = (const float*)d_in[4];
    float* out = (float*)d_out;

    const int nTok = in_sizes[0] / 4096;       // 4096 tokens
    int grid = 512;
    if (grid > nTok) grid = nTok;
    const int tpw = (nTok + grid - 1) / grid;  // 8 tokens per WG

    hipFuncSetAttribute((const void*)ttm_linear_kernel,
                        hipFuncAttributeMaxDynamicSharedMemorySize, SMEM_BYTES);
    ttm_linear_kernel<<<grid, 256, SMEM_BYTES, stream>>>(x, bias, core0, core1, core2,
                                                         out, tpw, nTok);
}
